// SelfAttention_45114336477480
// MI455X (gfx1250) — matrix operand order repaired and checked by reading, not compiled, not run on hardware
//
#include <hip/hip_runtime.h>
#include <hip/hip_bf16.h>

typedef __attribute__((ext_vector_type(16))) _Float16 v16h;
typedef __attribute__((ext_vector_type(8)))  _Float16 v8h;
typedef __attribute__((ext_vector_type(8)))  float    v8f;

#define EMBED_DIM 2048
#define NUM_HEADS 16
#define HEAD_DIM  128
#define BATCH     2
#define SEQ       2048

// ---------------------------------------------------------------------------
// gfx1250 async global->LDS copies (ASYNCcnt path), guarded so the file still
// compiles (with register-staging fallback) if this toolchain lacks them.
#ifndef __has_builtin
#define __has_builtin(x) 0
#endif
#if __has_builtin(__builtin_amdgcn_global_load_async_to_lds_b128) && \
    __has_builtin(__builtin_amdgcn_s_wait_asynccnt)
#define HAVE_ASYNC_LDS 1
#else
#define HAVE_ASYNC_LDS 0
#endif

#if HAVE_ASYNC_LDS
typedef int __attribute__((ext_vector_type(4))) int4v;
typedef __attribute__((address_space(1))) int4v* g_int4p;
typedef __attribute__((address_space(3))) int4v* l_int4p;
__device__ inline void async_copy_b128(const void* g, void* l) {
    __builtin_amdgcn_global_load_async_to_lds_b128((g_int4p)g, (l_int4p)l, 0, 0);
}
__device__ inline void wait_async0() { __builtin_amdgcn_s_wait_asynccnt(0); }
#else
__device__ inline void wait_async0() {}
#endif

// ---------------------------------------------------------------------------
// Fragment loaders per CDNA5 ISA 7.12.2 (wave32).
// A-matrix 16x32 f16 (MxK): lanes 0-15 -> rows, halves 0-7 = K[base..base+7],
// halves 8-15 = K[base+16..base+23], base = 0 (lanes 0-15) or 8 (lanes 16-31).
__device__ inline v16h load_fragA(const _Float16* base, int stride, int lane) {
    int r  = lane & 15;
    int kb = (lane >> 4) << 3;           // 0 or 8
    const _Float16* p = base + r * stride + kb;
    v8h lo = *(const v8h*)(p);
    v8h hi = *(const v8h*)(p + 16);
    v16h a;
#pragma unroll
    for (int i = 0; i < 8; ++i) { a[i] = lo[i]; a[8 + i] = hi[i]; }
    return a;
}

// B-matrix 32x16 f16 (KxN), accessed via Bt[n][k] (row-major rows = N):
// halves 0-7 = K[base..base+7], halves 8-15 = K[base+16..base+23], base = 0/8 per lane half, col = lane&15.
__device__ inline v16h load_fragB(const _Float16* base, int stride, int lane) {
    int c  = lane & 15;
    int kb = (lane >> 4) << 3;           // 0 or 8
    const _Float16* p = base + c * stride + kb;
    v8h lo = *(const v8h*)(p);
    v8h hi = *(const v8h*)(p + 16);
    v16h b;
#pragma unroll
    for (int i = 0; i < 8; ++i) { b[i] = lo[i]; b[8 + i] = hi[i]; }
    return b;
}

__device__ inline v8f wmma16x16x32(v16h a, v16h b, v8f c) {
    return __builtin_amdgcn_wmma_f32_16x16x32_f16(false, a, false, b,
                                                  (short)0, c, false, false);
}

// ---------------------------------------------------------------------------
// Y[m][n] = sum_k A[m][k] * W[n][k] + bias[n]     (i.e. A @ W^T + b)
// Block tile 128x128, 8 waves (wave w owns rows w*16..w*16+15), K chunk = 32.
// Double-buffered LDS; f16 A tiles go global->LDS via async DMA when available,
// fp32 tiles take the register stage (needed for the f32->f16 convert).
#define BM 128
#define BN 128
#define KC 32

template<int A_F16, int Y_F16>
__global__ __launch_bounds__(256)
void gemm_bias_kernel(const void* __restrict__ Aptr,
                      const float* __restrict__ W, const float* __restrict__ bias,
                      void* __restrict__ Yptr,
                      int M, int N, int K) {
    __shared__ __align__(16) _Float16 As[2][BM][KC];   // 16 KB
    __shared__ __align__(16) _Float16 Bs[2][BN][KC];   // 16 KB

    const int tid  = threadIdx.x;
    const int lane = tid & 31;
    const int wave = tid >> 5;
    const int bm   = blockIdx.y * BM;
    const int bn   = blockIdx.x * BN;
    const int row  = tid >> 1;            // 0..127 (2 threads/row)
    const int cs   = (tid & 1) * 16;      // 0 or 16

    const float*    Af32 = (const float*)Aptr;
    const _Float16* Af16 = (const _Float16*)Aptr;

    v8f acc[8];
#pragma unroll
    for (int i = 0; i < 8; ++i) acc[i] = (v8f){0,0,0,0,0,0,0,0};

    // Register stage for the next K chunk (fallback / fp32 paths).
    float af[16]; v8h ah0, ah1; float bf[16];

    auto load_global = [&](int k0, int pb) {
        if (A_F16) {
            const _Float16* src = Af16 + (size_t)(bm + row) * K + k0 + cs;
#if HAVE_ASYNC_LDS
            async_copy_b128(src,     &As[pb][row][cs]);
            async_copy_b128(src + 8, &As[pb][row][cs + 8]);
#else
            ah0 = *(const v8h*)(src);
            ah1 = *(const v8h*)(src + 8);
#endif
        } else {
            const float* src = Af32 + (size_t)(bm + row) * K + k0 + cs;
#pragma unroll
            for (int i = 0; i < 16; ++i) af[i] = src[i];
        }
        const float* wsrc = W + (size_t)(bn + row) * K + k0 + cs;
#pragma unroll
        for (int i = 0; i < 16; ++i) bf[i] = wsrc[i];
        (void)pb;
    };
    auto store_lds = [&](int pb) {
        if (A_F16) {
#if !HAVE_ASYNC_LDS
            *(v8h*)&As[pb][row][cs]     = ah0;
            *(v8h*)&As[pb][row][cs + 8] = ah1;
#endif
        } else {
#pragma unroll
            for (int i = 0; i < 16; ++i) As[pb][row][cs + i] = (_Float16)af[i];
        }
#pragma unroll
        for (int i = 0; i < 16; ++i) Bs[pb][row][cs + i] = (_Float16)bf[i];
    };

    load_global(0, 0);
    int p = 0;
    for (int k0 = 0; k0 < K; k0 += KC) {
        store_lds(p);
        if (A_F16) wait_async0();           // f16 A tile arrived in LDS
        __syncthreads();
        if (k0 + KC < K) load_global(k0 + KC, p ^ 1);  // overlaps WMMAs below

        v16h a = load_fragA(&As[p][wave * 16][0], KC, lane);
        v16h bfrag[8];
#pragma unroll
        for (int nt = 0; nt < 8; ++nt)
            bfrag[nt] = load_fragB(&Bs[p][nt * 16][0], KC, lane);
#pragma unroll
        for (int nt = 0; nt < 8; ++nt)
            acc[nt] = wmma16x16x32(a, bfrag[nt], acc[nt]);
        p ^= 1;
    }

    // Epilogue: C layout -> lane&15 = N, vgpr r = M (+8 for upper lane half).
    const int nl = lane & 15;
    const int mh = (lane >> 4) * 8;
#pragma unroll
    for (int nt = 0; nt < 8; ++nt) {
        int n = bn + nt * 16 + nl;
        float bv = bias[n];
#pragma unroll
        for (int r = 0; r < 8; ++r) {
            int m = bm + wave * 16 + mh + r;
            float val = acc[nt][r] + bv;
            if (Y_F16) ((_Float16*)Yptr)[(size_t)m * N + n] = (_Float16)val;
            else       ((float*)Yptr)[(size_t)m * N + n]    = val;
        }
    }
}

// ---------------------------------------------------------------------------
// RoPE (rotate-half form of the reference): pair (2j, 2j+1) within each head ->
// out[j] = e*cos - o*sin ; out[hd/2 + j] = o.   fp32 in, f16 planar out.
__global__ void rope_kernel(const float* __restrict__ Yq, const float* __restrict__ Yk,
                            _Float16* __restrict__ Qh, _Float16* __restrict__ Kh,
                            int BS, int S, int D, int hd) {
    int idx = blockIdx.x * blockDim.x + threadIdx.x;
    int total = BS * (D / 2);
    if (idx >= total) return;
    int row = idx / (D / 2);
    int p   = idx % (D / 2);
    int h   = p / (hd / 2);
    int j   = p % (hd / 2);
    int s   = row % S;

    float theta = __powf(10000.0f, -2.0f * (float)j / (float)hd);
    float fr = (float)s * theta;
    float cs = __cosf(fr), sn = __sinf(fr);

    size_t base = (size_t)row * D + (size_t)h * hd;
    float qe = Yq[base + 2 * j], qo = Yq[base + 2 * j + 1];
    float ke = Yk[base + 2 * j], ko = Yk[base + 2 * j + 1];
    Qh[base + j]          = (_Float16)(qe * cs - qo * sn);
    Qh[base + hd / 2 + j] = (_Float16)qo;
    Kh[base + j]          = (_Float16)(ke * cs - ko * sn);
    Kh[base + hd / 2 + j] = (_Float16)ko;
}

// ---------------------------------------------------------------------------
// Flash attention. Block = 8 waves = 128 query rows, each wave owns 16 rows.
// Keys stream global->LDS via async DMA (when available); values take the
// register stage because staging transposes them. Double-buffered, online
// softmax kept in accumulator registers.
#define BC 32

__global__ __launch_bounds__(256)
void flash_attn_kernel(const _Float16* __restrict__ Q, const _Float16* __restrict__ Kp,
                       const _Float16* __restrict__ V, _Float16* __restrict__ Ctx,
                       int B, int H, int S, int D) {
    __shared__ __align__(16) _Float16 Ks[2][BC][HEAD_DIM];   // 16 KB (keys, row-major)
    __shared__ __align__(16) _Float16 Vt[2][HEAD_DIM][BC];   // 16 KB (values, transposed)
    __shared__ __align__(16) _Float16 Ps[8][16][BC];         //  8 KB (per-wave P scratch)

    const int tid  = threadIdx.x;
    const int lane = tid & 31;
    const int wave = tid >> 5;
    const int bh   = blockIdx.y;
    const int b    = bh / H;
    const int h    = bh % H;
    const int q0   = blockIdx.x * 128 + wave * 16;
    const size_t rowbase = (size_t)b * S * D + (size_t)h * HEAD_DIM;
    const float scale = 0.08838834764831845f;   // 1/sqrt(128)

    // This wave's Q tile as 4 A-fragments (hd = 4 * 32), pre-scaled by 1/sqrt(hd).
    v16h qf[4];
    {
        int r  = lane & 15;
        int kb = (lane >> 4) << 3;
        const _Float16* qp = Q + rowbase + (size_t)(q0 + r) * D;
#pragma unroll
        for (int c = 0; c < 4; ++c) {
            v8h lo = *(const v8h*)(qp + c * 32 + kb);
            v8h hi = *(const v8h*)(qp + c * 32 + kb + 16);
#pragma unroll
            for (int i = 0; i < 8; ++i) {
                qf[c][i]     = (_Float16)((float)lo[i] * scale);
                qf[c][8 + i] = (_Float16)((float)hi[i] * scale);
            }
        }
    }

    float mrow[8], lrow[8];
    v8f o[8];
#pragma unroll
    for (int i = 0; i < 8; ++i) { mrow[i] = -1e30f; lrow[i] = 0.0f; o[i] = (v8f){0,0,0,0,0,0,0,0}; }

    // Staging: 8 threads/row, 16 halves each.
    const int srow = tid >> 3;
    const int scs  = (tid & 7) * 16;
    v8h vreg0, vreg1;
#if !HAVE_ASYNC_LDS
    v8h kreg0, kreg1;
#endif

    auto issue_loads = [&](int kv0, int pb) {
        const _Float16* vp = V + rowbase + (size_t)(kv0 + srow) * D + scs;
        vreg0 = *(const v8h*)(vp);
        vreg1 = *(const v8h*)(vp + 8);
        const _Float16* kp = Kp + rowbase + (size_t)(kv0 + srow) * D + scs;
#if HAVE_ASYNC_LDS
        async_copy_b128(kp,     &Ks[pb][srow][scs]);
        async_copy_b128(kp + 8, &Ks[pb][srow][scs + 8]);
#else
        kreg0 = *(const v8h*)(kp);
        kreg1 = *(const v8h*)(kp + 8);
#endif
        (void)pb;
    };
    auto stage_lds = [&](int pb) {
#if !HAVE_ASYNC_LDS
        *(v8h*)&Ks[pb][srow][scs]     = kreg0;
        *(v8h*)&Ks[pb][srow][scs + 8] = kreg1;
#endif
#pragma unroll
        for (int i = 0; i < 8; ++i) {
            Vt[pb][scs + i][srow]     = vreg0[i];
            Vt[pb][scs + 8 + i][srow] = vreg1[i];
        }
    };

    issue_loads(0, 0);
    int p = 0;
    for (int kv0 = 0; kv0 < S; kv0 += BC) {
        stage_lds(p);
        wait_async0();                          // K tile for buffer p landed
        __syncthreads();
        if (kv0 + BC < S) issue_loads(kv0 + BC, p ^ 1);  // overlaps compute

        // Scores: 16 x 32 per wave, two 16-col C tiles, K-reduce over hd.
        v16h kfrag[8];
#pragma unroll
        for (int nt = 0; nt < 2; ++nt)
#pragma unroll
            for (int kc = 0; kc < 4; ++kc)
                kfrag[nt * 4 + kc] = load_fragB(&Ks[p][nt * 16][kc * 32], HEAD_DIM, lane);
        v8f sc[2];
#pragma unroll
        for (int nt = 0; nt < 2; ++nt) {
            v8f c = (v8f){0,0,0,0,0,0,0,0};
#pragma unroll
            for (int kc = 0; kc < 4; ++kc)
                c = wmma16x16x32(qf[kc], kfrag[nt * 4 + kc], c);
            sc[nt] = c;
        }

        // Online softmax: rows live across lanes 0-15 (or 16-31) of one VGPR.
#pragma unroll
        for (int r = 0; r < 8; ++r) {
            float s0 = sc[0][r];
            float s1 = sc[1][r];
            float v  = fmaxf(s0, s1);
#pragma unroll
            for (int m = 1; m < 16; m <<= 1) v = fmaxf(v, __shfl_xor(v, m, 32));
            float mnew  = fmaxf(mrow[r], v);
            float alpha = __expf(mrow[r] - mnew);
            mrow[r] = mnew;
            s0 = __expf(s0 - mnew);
            s1 = __expf(s1 - mnew);
            sc[0][r] = s0; sc[1][r] = s1;
            float rs = s0 + s1;
#pragma unroll
            for (int m = 1; m < 16; m <<= 1) rs += __shfl_xor(rs, m, 32);
            lrow[r] = lrow[r] * alpha + rs;
#pragma unroll
            for (int nt = 0; nt < 8; ++nt) o[nt][r] *= alpha;
        }

        // P tile: C layout -> LDS (row-major 16x32) -> A fragment.
        {
            int nl = lane & 15, mh = (lane >> 4) * 8;
#pragma unroll
            for (int nt = 0; nt < 2; ++nt)
#pragma unroll
                for (int r = 0; r < 8; ++r)
                    Ps[wave][mh + r][nt * 16 + nl] = (_Float16)sc[nt][r];
        }
        __syncthreads();

        v16h pf = load_fragA(&Ps[wave][0][0], BC, lane);
        v16h vfrag[8];
#pragma unroll
        for (int nt = 0; nt < 8; ++nt)
            vfrag[nt] = load_fragB(&Vt[p][nt * 16][0], BC, lane);
#pragma unroll
        for (int nt = 0; nt < 8; ++nt)
            o[nt] = wmma16x16x32(pf, vfrag[nt], o[nt]);
        p ^= 1;
    }

    // Normalize and store ctx in planar (B,S,D) f16.
    {
        int nl = lane & 15, mh = (lane >> 4) * 8;
#pragma unroll
        for (int nt = 0; nt < 8; ++nt)
#pragma unroll
            for (int r = 0; r < 8; ++r) {
                float val = o[nt][r] / lrow[r];
                int m = q0 + mh + r;
                Ctx[(size_t)(b * S + m) * D + h * HEAD_DIM + nt * 16 + nl] = (_Float16)val;
            }
    }
}

// ---------------------------------------------------------------------------
extern "C" void kernel_launch(void* const* d_in, const int* in_sizes, int n_in,
                              void* d_out, int out_size, void* d_ws, size_t ws_size,
                              hipStream_t stream) {
    (void)in_sizes; (void)n_in; (void)out_size; (void)ws_size;
    const float* x  = (const float*)d_in[0];
    const float* wq = (const float*)d_in[1];
    const float* bq = (const float*)d_in[2];
    const float* wk = (const float*)d_in[3];
    const float* bk = (const float*)d_in[4];
    const float* wv = (const float*)d_in[5];
    const float* bv = (const float*)d_in[6];
    const float* wo = (const float*)d_in[7];
    const float* bo = (const float*)d_in[8];

    const int B = BATCH, S = SEQ, D = EMBED_DIM, H = NUM_HEADS;
    const int M = B * S;                       // 4096

    char* ws = (char*)d_ws;
    float*    Yq = (float*)(ws);                              // 33.5 MB fp32 pre-RoPE Q
    float*    Yk = (float*)(ws + (size_t)33554432);           // 33.5 MB fp32 pre-RoPE K
    _Float16* Qh = (_Float16*)(ws + (size_t)67108864);        // 16.8 MB f16 Q (post-RoPE)
    _Float16* Kh = (_Float16*)(ws + (size_t)83886080);        // 16.8 MB f16 K (post-RoPE)
    _Float16* Vh = (_Float16*)(ws + (size_t)100663296);       // 16.8 MB f16 V
    _Float16* Ct = (_Float16*)(ws + (size_t)117440512);       // 16.8 MB f16 ctx

    dim3 gg(D / BN, M / BM);                   // (16, 32)
    gemm_bias_kernel<0, 0><<<gg, 256, 0, stream>>>(x, wq, bq, Yq, M, D, D);
    gemm_bias_kernel<0, 0><<<gg, 256, 0, stream>>>(x, wk, bk, Yk, M, D, D);
    gemm_bias_kernel<0, 1><<<gg, 256, 0, stream>>>(x, wv, bv, Vh, M, D, D);

    int pairs = M * (D / 2);
    rope_kernel<<<(pairs + 255) / 256, 256, 0, stream>>>(Yq, Yk, Qh, Kh, M, S, D, HEAD_DIM);

    dim3 ga(S / 128, B * H);                   // (16, 32)
    flash_attn_kernel<<<ga, 256, 0, stream>>>(Qh, Kh, Vh, Ct, B, H, S, D);

    gemm_bias_kernel<1, 0><<<gg, 256, 0, stream>>>(Ct, wo, bo, d_out, M, D, D);
}
